// DGCNN_Seg_74826920231263
// MI455X (gfx1250) — compile-verified, matrix operand order checked by reading, not run on hardware
//
#include <hip/hip_runtime.h>
#include <hip/hip_bf16.h>

// ---------------------------------------------------------------------------
// Types / helpers
// ---------------------------------------------------------------------------
typedef __attribute__((ext_vector_type(16))) _Float16 v16h;
typedef __attribute__((ext_vector_type(8)))  float    v8f;

#define DEV __device__ __forceinline__

union F16x16 { v16h v; _Float16 e[16]; };

// A 16x32 f16 fragment (per ISA 7.12.2): lane L, g=L>>4, r=L&15 (row for A /
// col for B). halves 0..7 -> K = k0 + g*8 + h (contiguous), halves 8..15 ->
// K = k0 + 16 + g*8 + (h-8) (contiguous).  Two 16B contiguous loads.
DEV v16h load_frag(const _Float16* base, int pitch, int rc, int k0, int g) {
    const _Float16* p = base + (size_t)rc * pitch + k0 + g * 8;
    F16x16 u;
#pragma unroll
    for (int i = 0; i < 8; ++i) u.e[i] = p[i];
#pragma unroll
    for (int i = 0; i < 8; ++i) u.e[8 + i] = p[16 + i];
    return u.v;
}

// monotone float <-> uint mapping for atomic max on floats
DEV unsigned f2ord(float f) {
    unsigned u = __float_as_uint(f);
    return (u & 0x80000000u) ? ~u : (u | 0x80000000u);
}
DEV float ord2f(unsigned o) {
    return (o & 0x80000000u) ? __uint_as_float(o & 0x7fffffffu) : __uint_as_float(~o);
}

DEV float lrelu(float y) { return y > 0.f ? y : 0.2f * y; }

static constexpr int NPTS = 4096;   // points per batch
static constexpr int BATCH = 2;
static constexpr int KNN  = 20;
static constexpr int MROWS = BATCH * NPTS;  // 8192

// ---------------------------------------------------------------------------
// squared norms per point
// ---------------------------------------------------------------------------
__global__ void sqnorm_kernel(const float* __restrict__ x, int ld, int C,
                              float* __restrict__ xx, int total) {
    int r = blockIdx.x * blockDim.x + threadIdx.x;
    if (r >= total) return;
    const float* p = x + (size_t)r * ld;
    float s = 0.f;
    for (int c = 0; c < C; ++c) { float v = p[c]; s += v * v; }
    xx[r] = s;
}

// ---------------------------------------------------------------------------
// kNN (20 smallest of d = |xn|^2 + |xm|^2 - 2 xn.xm), exact f32.
// One wave per query row; lane-strided candidates; per-lane sorted top-20 in
// registers; 640-candidate merge in LDS with shfl min-reduction.
// ---------------------------------------------------------------------------
template <int C, int LDX>
__global__ void knn_kernel(const float* __restrict__ xin,
                           const float* __restrict__ xx,
                           int* __restrict__ knn) {
    constexpr int RB = (C < 4) ? 4 : C;
    __shared__ __align__(16) float rowbuf[8][RB];
    __shared__ float cdist[8][32 * KNN];
    __shared__ int   cidx [8][32 * KNN];

    const int wave = threadIdx.x >> 5;
    const int lane = threadIdx.x & 31;
    const int grow = blockIdx.x * 8 + wave;          // global row
    const int b = grow >> 12;
    const size_t bbase = (size_t)b * NPTS;

    const float* xr = xin + (size_t)grow * LDX;
    for (int c = lane; c < C; c += 32) rowbuf[wave][c] = xr[c];
    __syncthreads();

    const float xn = xx[grow];
    float bd[KNN]; int bi[KNN];
#pragma unroll
    for (int j = 0; j < KNN; ++j) { bd[j] = 3.0e38f; bi[j] = 0; }

    for (int t = 0; t < NPTS / 32; ++t) {
        const int m = lane + t * 32;
        const float* xm = xin + (bbase + m) * LDX;
        float dot = 0.f;
        if constexpr (C == 3) {
            dot = rowbuf[wave][0] * xm[0] + rowbuf[wave][1] * xm[1] +
                  rowbuf[wave][2] * xm[2];
        } else {
#pragma unroll 4
            for (int c = 0; c < C; c += 4) {
                const float4 mv = *(const float4*)(xm + c);
                const float4 rv = *(const float4*)(&rowbuf[wave][c]);
                dot += rv.x * mv.x + rv.y * mv.y + rv.z * mv.z + rv.w * mv.w;
            }
        }
        const float d = xn + xx[bbase + m] - 2.0f * dot;
        if (d < bd[KNN - 1]) {           // sorted insertion (unrolled)
            float dd = d; int ii = m;
#pragma unroll
            for (int j = 0; j < KNN; ++j) {
                if (dd < bd[j]) {
                    float td = bd[j]; int ti = bi[j];
                    bd[j] = dd; bi[j] = ii; dd = td; ii = ti;
                }
            }
        }
    }
#pragma unroll
    for (int j = 0; j < KNN; ++j) {
        cdist[wave][lane * KNN + j] = bd[j];
        cidx [wave][lane * KNN + j] = bi[j];
    }
    __syncthreads();

    for (int r = 0; r < KNN; ++r) {
        float v = 3.0e38f; int pos = 0;
#pragma unroll
        for (int j = 0; j < KNN; ++j) {
            const float cv = cdist[wave][lane * KNN + j];
            if (cv < v) { v = cv; pos = lane * KNN + j; }
        }
        for (int off = 16; off; off >>= 1) {
            const float ov = __shfl_xor(v, off, 32);
            const int   op = __shfl_xor(pos, off, 32);
            if (ov < v || (ov == v && op < pos)) { v = ov; pos = op; }
        }
        if (lane == 0) {
            knn[(size_t)grow * KNN + r] = cidx[wave][pos];
            cdist[wave][pos] = 3.0e38f;
        }
        __syncthreads();
    }
}

// ---------------------------------------------------------------------------
// EdgeConv: for 16 points, loop k over 20 neighbors; per k build feature tile
// [16 x 2C] = [nbr-xc, xc] in LDS (f16), WMMA against register-cached weight
// fragments, fused scale/bias/lrelu + running max over k.
// ---------------------------------------------------------------------------
template <int CIN, int COUT, int NW>
__global__ void edge_conv_kernel(const float* __restrict__ xin, int ldx,
                                 const int* __restrict__ knn,
                                 const float* __restrict__ W,
                                 const float* __restrict__ S,
                                 const float* __restrict__ Bi,
                                 float* __restrict__ out, int ldo) {
    constexpr int TWOC = 2 * CIN;
    constexpr int CH   = (TWOC + 31) / 32;   // K-chunks of 32
    constexpr int CHP  = CH * 32;            // padded 2C
    constexpr int NT   = NW * 32;
    constexpr int TPW  = COUT / 16 / NW;     // column tiles per wave

    __shared__ __align__(16) _Float16 Alds[16 * CHP];

    const int tid  = threadIdx.x;
    const int wave = tid >> 5;
    const int lane = tid & 31;
    const int g    = lane >> 4;
    const int hn   = lane & 15;
    const int p0   = blockIdx.x * 16;

    // cache B fragments (weights, constant across k and points) + scale/bias
    F16x16 bc[TPW][CH];
    float sc[TPW], bb[TPW];
#pragma unroll
    for (int t = 0; t < TPW; ++t) {
        const int o = (wave * TPW + t) * 16 + hn;
        sc[t] = S[o]; bb[t] = Bi[o];
#pragma unroll
        for (int kc = 0; kc < CH; ++kc) {
#pragma unroll
            for (int h = 0; h < 16; ++h) {
                const int kk = kc * 32 + (h < 8 ? g * 8 + h : 16 + g * 8 + (h - 8));
                bc[t][kc].e[h] = (kk < TWOC) ? (_Float16)W[(size_t)o * TWOC + kk]
                                             : (_Float16)0.f;
            }
        }
    }

    float rmax[TPW][8];
#pragma unroll
    for (int t = 0; t < TPW; ++t)
#pragma unroll
        for (int j = 0; j < 8; ++j) rmax[t][j] = -3.0e38f;

    for (int k = 0; k < KNN; ++k) {
        for (int e = tid; e < 16 * CHP; e += NT) {
            const int p = e / CHP, c = e - p * CHP;
            const int grow = p0 + p;
            float v = 0.f;
            if (c < TWOC) {
                if (c < CIN) {
                    const int b  = grow >> 12;
                    const int nb = knn[(size_t)grow * KNN + k];
                    v = xin[(size_t)((b << 12) + nb) * ldx + c] -
                        xin[(size_t)grow * ldx + c];
                } else {
                    v = xin[(size_t)grow * ldx + (c - CIN)];
                }
            }
            Alds[p * CHP + c] = (_Float16)v;
        }
        __syncthreads();

        v8f acc[TPW];
#pragma unroll
        for (int t = 0; t < TPW; ++t)
#pragma unroll
            for (int j = 0; j < 8; ++j) acc[t][j] = 0.f;

#pragma unroll
        for (int kc = 0; kc < CH; ++kc) {
            const v16h af = load_frag(Alds, CHP, hn, kc * 32, g);
#pragma unroll
            for (int t = 0; t < TPW; ++t)
                acc[t] = __builtin_amdgcn_wmma_f32_16x16x32_f16(
                    false, af, false, bc[t][kc].v, (short)0, acc[t], false, false);
        }
#pragma unroll
        for (int t = 0; t < TPW; ++t)
#pragma unroll
            for (int j = 0; j < 8; ++j) {
                const float y = lrelu(acc[t][j] * sc[t] + bb[t]);
                rmax[t][j] = fmaxf(rmax[t][j], y);
            }
        __syncthreads();
    }
#pragma unroll
    for (int t = 0; t < TPW; ++t) {
        const int o0 = (wave * TPW + t) * 16;
#pragma unroll
        for (int j = 0; j < 8; ++j) {
            const int grow = p0 + g * 8 + j;
            out[(size_t)grow * ldo + o0 + hn] = rmax[t][j];
        }
    }
}

// ---------------------------------------------------------------------------
// Generic WMMA GEMM:  y[m,o] = (sum_c A[m,c]*W[o,c] + gterm[b,o]) * S[o] + Bi[o]
// optional LeakyReLU; MAXOUT variant reduces max over rows into gord (ordered
// uint atomics) instead of storing rows (used for the global feature g).
// Block: 256 thr = 8 waves -> 32x64 tile, K staged in 32-chunks via LDS.
// ---------------------------------------------------------------------------
template <bool LRELU, bool MAXOUT>
__global__ void gemm_kernel(const float* __restrict__ A, int lda, int kdim,
                            const float* __restrict__ W, int ldw, int cout,
                            const float* __restrict__ S,
                            const float* __restrict__ Bi,
                            const float* __restrict__ gterm, int gtld,
                            float* __restrict__ out, int ldo,
                            unsigned* __restrict__ gord) {
    __shared__ __align__(16) _Float16 Alds[32 * 32];
    __shared__ __align__(16) _Float16 Blds[64 * 32];

    const int tid  = threadIdx.x;
    const int wave = tid >> 5;
    const int lane = tid & 31;
    const int g    = lane >> 4;
    const int hn   = lane & 15;
    const int r2   = wave >> 2;      // row half (0/1)
    const int colt = wave & 3;       // column tile (0..3)
    const int m0   = blockIdx.x * 32;
    const int o0b  = blockIdx.y * 64;

    v8f acc;
#pragma unroll
    for (int j = 0; j < 8; ++j) acc[j] = 0.f;

    for (int kc0 = 0; kc0 < kdim; kc0 += 32) {
        {   // A tile 32x32 (f32 -> f16), one float4 per thread
            const int row = tid >> 3, q = (tid & 7) * 4;
            const float4 v = *(const float4*)(A + (size_t)(m0 + row) * lda + kc0 + q);
            _Float16* d = Alds + row * 32 + q;
            d[0] = (_Float16)v.x; d[1] = (_Float16)v.y;
            d[2] = (_Float16)v.z; d[3] = (_Float16)v.w;
        }
#pragma unroll
        for (int qq = 0; qq < 2; ++qq) {   // B tile 64x32
            const int q = tid * 2 + qq;
            const int o = q >> 3, kp = (q & 7) * 4;
            _Float16* d = Blds + o * 32 + kp;
            const int og = o0b + o;
            if (og < cout) {
                const float4 v = *(const float4*)(W + (size_t)og * ldw + kc0 + kp);
                d[0] = (_Float16)v.x; d[1] = (_Float16)v.y;
                d[2] = (_Float16)v.z; d[3] = (_Float16)v.w;
            } else {
                d[0] = d[1] = d[2] = d[3] = (_Float16)0.f;
            }
        }
        __syncthreads();
        const v16h af = load_frag(Alds, 32, r2 * 16 + hn, 0, g);
        const v16h bf = load_frag(Blds, 32, colt * 16 + hn, 0, g);
        acc = __builtin_amdgcn_wmma_f32_16x16x32_f16(
            false, af, false, bf, (short)0, acc, false, false);
        __syncthreads();
    }

    const int o = o0b + colt * 16 + hn;
    const int b = m0 >> 12;
    if (o < cout) {
        const float scv = S ? S[o] : 1.0f;
        const float bv  = Bi[o];
        const float gt  = gterm ? gterm[b * gtld + o] : 0.f;
        if (MAXOUT) {
            float mx = -3.0e38f;
#pragma unroll
            for (int j = 0; j < 8; ++j) {
                float y = (acc[j] + gt) * scv + bv;
                if (LRELU) y = lrelu(y);
                mx = fmaxf(mx, y);
            }
            mx = fmaxf(mx, __shfl_xor(mx, 16, 32));
            if (g == 0) atomicMax(gord + b * cout + o, f2ord(mx));
        } else {
#pragma unroll
            for (int j = 0; j < 8; ++j) {
                float y = (acc[j] + gt) * scv + bv;
                if (LRELU) y = lrelu(y);
                out[(size_t)(m0 + r2 * 16 + g * 8 + j) * ldo + o] = y;
            }
        }
    }
}

// ---------------------------------------------------------------------------
// gterm[b,o] = sum_{c<1024} Ws1[o, 512+c] * g[b,c]   (one wave per output)
// ---------------------------------------------------------------------------
__global__ void gterm_kernel(const unsigned* __restrict__ gord,
                             const float* __restrict__ Ws1,
                             float* __restrict__ gterm) {
    const int w = (blockIdx.x * blockDim.x + threadIdx.x) >> 5;
    const int lane = threadIdx.x & 31;
    const int b = w >> 9, o = w & 511;
    float s = 0.f;
    for (int c = lane; c < 1024; c += 32)
        s += ord2f(gord[b * 1024 + c]) * Ws1[(size_t)o * 1536 + 512 + c];
    for (int off = 16; off; off >>= 1) s += __shfl_xor(s, off, 32);
    if (lane == 0) gterm[b * 512 + o] = s;
}

__global__ void init_gord(unsigned* g, int n) {
    const int i = blockIdx.x * blockDim.x + threadIdx.x;
    if (i < n) g[i] = 0u;   // ord-space -infinity sentinel
}

// ---------------------------------------------------------------------------
// launch
// ---------------------------------------------------------------------------
extern "C" void kernel_launch(void* const* d_in, const int* in_sizes, int n_in,
                              void* d_out, int out_size, void* d_ws, size_t ws_size,
                              hipStream_t stream) {
    const float* x   = (const float*)d_in[0];
    const float* W1  = (const float*)d_in[1];
    const float* s1  = (const float*)d_in[2];
    const float* b1  = (const float*)d_in[3];
    const float* W2  = (const float*)d_in[4];
    const float* s2  = (const float*)d_in[5];
    const float* b2  = (const float*)d_in[6];
    const float* W3  = (const float*)d_in[7];
    const float* s3  = (const float*)d_in[8];
    const float* b3  = (const float*)d_in[9];
    const float* W4  = (const float*)d_in[10];
    const float* s4  = (const float*)d_in[11];
    const float* b4  = (const float*)d_in[12];
    const float* Wg  = (const float*)d_in[13];
    const float* sg  = (const float*)d_in[14];
    const float* bg  = (const float*)d_in[15];
    const float* Ws1 = (const float*)d_in[16];
    const float* ss1 = (const float*)d_in[17];
    const float* bs1 = (const float*)d_in[18];
    const float* Ws2 = (const float*)d_in[19];
    const float* ss2 = (const float*)d_in[20];
    const float* bs2 = (const float*)d_in[21];
    const float* Ws3 = (const float*)d_in[22];
    const float* bs3 = (const float*)d_in[23];
    (void)in_sizes; (void)n_in; (void)out_size; (void)ws_size;

    char* ws = (char*)d_ws;
    constexpr size_t OFF_XX   = 0;                            // 32 KB
    constexpr size_t OFF_IDX  = OFF_XX   + (size_t)MROWS * 4;           // 640 KB
    constexpr size_t OFF_XCAT = OFF_IDX  + (size_t)MROWS * KNN * 4;     // 16 MB
    constexpr size_t OFF_GORD = OFF_XCAT + (size_t)MROWS * 512 * 4;     // 8 KB
    constexpr size_t OFF_GT   = OFF_GORD + (size_t)BATCH * 1024 * 4;    // 4 KB
    constexpr size_t OFF_H1   = OFF_GT   + (size_t)BATCH * 512 * 4;     // 16 MB
    constexpr size_t OFF_H2   = OFF_H1   + (size_t)MROWS * 512 * 4;     // 8 MB

    float*    xx    = (float*)(ws + OFF_XX);
    int*      knn   = (int*)  (ws + OFF_IDX);
    float*    xcat  = (float*)(ws + OFF_XCAT);
    unsigned* gord  = (unsigned*)(ws + OFF_GORD);
    float*    gterm = (float*)(ws + OFF_GT);
    float*    h1    = (float*)(ws + OFF_H1);
    float*    h2    = (float*)(ws + OFF_H2);
    float*    logits = (float*)d_out;

    const int M = MROWS;

    // ---- stage 1 (C=3 -> 64 @ xcat[:,0:64]) ----
    sqnorm_kernel<<<M / 256, 256, 0, stream>>>(x, 3, 3, xx, M);
    knn_kernel<3, 3><<<M / 8, 256, 0, stream>>>(x, xx, knn);
    edge_conv_kernel<3, 64, 4><<<M / 16, 128, 0, stream>>>(
        x, 3, knn, W1, s1, b1, xcat + 0, 512);

    // ---- stage 2 (64 -> 64 @ xcat[:,64:128]) ----
    sqnorm_kernel<<<M / 256, 256, 0, stream>>>(xcat + 0, 512, 64, xx, M);
    knn_kernel<64, 512><<<M / 8, 256, 0, stream>>>(xcat + 0, xx, knn);
    edge_conv_kernel<64, 64, 4><<<M / 16, 128, 0, stream>>>(
        xcat + 0, 512, knn, W2, s2, b2, xcat + 64, 512);

    // ---- stage 3 (64 -> 128 @ xcat[:,128:256]) ----
    sqnorm_kernel<<<M / 256, 256, 0, stream>>>(xcat + 64, 512, 64, xx, M);
    knn_kernel<64, 512><<<M / 8, 256, 0, stream>>>(xcat + 64, xx, knn);
    edge_conv_kernel<64, 128, 8><<<M / 16, 256, 0, stream>>>(
        xcat + 64, 512, knn, W3, s3, b3, xcat + 128, 512);

    // ---- stage 4 (128 -> 256 @ xcat[:,256:512]) ----
    sqnorm_kernel<<<M / 256, 256, 0, stream>>>(xcat + 128, 512, 128, xx, M);
    knn_kernel<128, 512><<<M / 8, 256, 0, stream>>>(xcat + 128, xx, knn);
    edge_conv_kernel<128, 256, 8><<<M / 16, 256, 0, stream>>>(
        xcat + 128, 512, knn, W4, s4, b4, xcat + 256, 512);

    // ---- g = max_n lrelu((xcat @ Wg^T) * sg + bg), fused atomic-max ----
    init_gord<<<(BATCH * 1024 + 255) / 256, 256, 0, stream>>>(gord, BATCH * 1024);
    gemm_kernel<true, true><<<dim3(M / 32, 16), 256, 0, stream>>>(
        xcat, 512, 512, Wg, 512, 1024, sg, bg, nullptr, 0, nullptr, 0, gord);

    // ---- gterm = Ws1[:,512:] . g  (folds the [xcat,g] concat) ----
    gterm_kernel<<<128, 256, 0, stream>>>(gord, Ws1, gterm);

    // ---- h1 = lrelu((xcat @ Ws1[:, :512]^T + gterm) * ss1 + bs1) ----
    gemm_kernel<true, false><<<dim3(M / 32, 8), 256, 0, stream>>>(
        xcat, 512, 512, Ws1, 1536, 512, ss1, bs1, gterm, 512, h1, 512, nullptr);

    // ---- h2 = lrelu((h1 @ Ws2^T) * ss2 + bs2) ----
    gemm_kernel<true, false><<<dim3(M / 32, 4), 256, 0, stream>>>(
        h1, 512, 512, Ws2, 512, 256, ss2, bs2, nullptr, 0, h2, 256, nullptr);

    // ---- logits = h2 @ Ws3^T + bs3 ----
    gemm_kernel<false, false><<<dim3(M / 32, 1), 256, 0, stream>>>(
        h2, 256, 256, Ws3, 256, 13, nullptr, bs3, nullptr, 0, logits, 13, nullptr);
}
